// CFDError_77541339562372
// MI455X (gfx1250) — matrix operand order repaired and checked by reading, not compiled, not run on hardware
//
#include <hip/hip_runtime.h>

// -------- CDNA5 WMMA / vector types ----------
typedef __attribute__((ext_vector_type(16))) _Float16 v16h;
typedef __attribute__((ext_vector_type(8)))  _Float16 h8;
typedef __attribute__((ext_vector_type(8)))  float    v8f;
typedef __attribute__((ext_vector_type(4)))  float    f32x4;

#define WAVES_PER_BLOCK 4
#define EDGE_BLOCK 128
#define BN_EPS 1e-5f

__device__ __forceinline__ v8f wmma_f16(v16h a, v16h b, v8f c) {
  return __builtin_amdgcn_wmma_f32_16x16x32_f16(false, a, false, b, (short)0, c,
                                                false, false);
}
// Force a wave-uniform value into an SGPR so guards become scalar branches.
__device__ __forceinline__ int uni(int v) {
  return __builtin_amdgcn_readfirstlane(v);
}
__device__ __forceinline__ f32x4 ld4(const float* p) { return *(const f32x4*)p; }
__device__ __forceinline__ v8f zero8f() {
  v8f r;
#pragma unroll
  for (int i = 0; i < 8; ++i) r[i] = 0.f;
  return r;
}
__device__ __forceinline__ v16h zero16h() {
  v16h r;
#pragma unroll
  for (int i = 0; i < 16; ++i) r[i] = (_Float16)0.f;
  return r;
}
__device__ __forceinline__ v16h pack16(f32x4 a0, f32x4 a1, f32x4 a2, f32x4 a3) {
  v16h r;
#pragma unroll
  for (int i = 0; i < 4; ++i) {
    r[i]      = (_Float16)a0[i];
    r[4 + i]  = (_Float16)a1[i];
    r[8 + i]  = (_Float16)a2[i];
    r[12 + i] = (_Float16)a3[i];
  }
  return r;
}
// two 8-half runs at p and p+16 (two b128 LDS loads) -> B fragment
__device__ __forceinline__ v16h ldb16(const _Float16* p) {
  h8 lo = *(const h8*)p;
  h8 hi = *(const h8*)(p + 16);
  v16h r;
#pragma unroll
  for (int i = 0; i < 8; ++i) { r[i] = lo[i]; r[8 + i] = hi[i]; }
  return r;
}
__device__ __forceinline__ unsigned packh2(float x, float y) {
  union { _Float16 h[2]; unsigned u; } r;
  r.h[0] = (_Float16)x; r.h[1] = (_Float16)y;
  return r.u;
}

// Hardware tanh (CDNA5 TRANS32 op).
__device__ __forceinline__ float fast_tanh(float x) {
#if __has_builtin(__builtin_amdgcn_tanhf)
  return __builtin_amdgcn_tanhf(x);
#elif __has_builtin(__builtin_amdgcn_tanh_f32)
  return __builtin_amdgcn_tanh_f32(x);
#else
  float r;
  asm volatile("v_tanh_f32 %0, %1\n\tv_nop" : "=v"(r) : "v"(x));
  return r;
#endif
}

// Per-edge activations are stored channel-interleaved, 32 halves per edge:
// slot 2c -> channel c (c<16), slot 2c+1 -> channel 16+c.
// A-fragment for lane (hi = lane>=16): 16 contiguous halves at offset 16*hi.
__device__ __forceinline__ v16h load_bn_a(const _Float16* __restrict__ hrow,
                                          const float* __restrict__ ab, int hi) {
  int cb0 = hi * 8;        // channels for even slots (K = cb0+j)
  int cb1 = 16 + hi * 8;   // channels for odd slots  (K = cb1+j)
  h8 r0 = *(const h8*)(hrow + 16 * hi);
  h8 r1 = *(const h8*)(hrow + 16 * hi + 8);
  f32x4 sc0a = ld4(ab + cb0),      sc0b = ld4(ab + cb0 + 4);
  f32x4 sc1a = ld4(ab + cb1),      sc1b = ld4(ab + cb1 + 4);
  f32x4 sh0a = ld4(ab + 32 + cb0), sh0b = ld4(ab + 32 + cb0 + 4);
  f32x4 sh1a = ld4(ab + 32 + cb1), sh1b = ld4(ab + 32 + cb1 + 4);
  v16h a;
#pragma unroll
  for (int j = 0; j < 8; ++j) {
    int pe = 2 * j, po = 2 * j + 1;
    float ve = (float)((pe < 8) ? r0[pe] : r1[pe - 8]);
    float vo = (float)((po < 8) ? r0[po] : r1[po - 8]);
    float se = (j < 4) ? sc0a[j] : sc0b[j - 4];
    float te = (j < 4) ? sh0a[j] : sh0b[j - 4];
    float so = (j < 4) ? sc1a[j] : sc1b[j - 4];
    float to = (j < 4) ? sh1a[j] : sh1b[j - 4];
    a[j]     = (_Float16)(se * ve + te);
    a[8 + j] = (_Float16)(so * vo + to);
  }
  return a;
}

// Shared tail for edge layers 1/2: bias+tanh+interleaved store+stats.
// e0 is an SGPR value (from uni()), so the full/partial split is a scalar
// branch, and the fast path has constant-immediate store offsets.
__device__ __forceinline__ void edge_act_store(
    v8f c0, v8f c1, float bias0, float bias1, _Float16* __restrict__ hout,
    int e0, int hi, int n, int E, float* s_sum, float* s_sq) {
  float sum0 = 0.f, sq0 = 0.f, sum1 = 0.f, sq1 = 0.f;
  _Float16* hb = hout + (long long)e0 * 32 + hi * 256 + 2 * n;
  if (e0 + 16 <= E) {                       // scalar branch (e0 uniform)
#pragma unroll
    for (int r = 0; r < 8; ++r) {
      float v0 = fast_tanh(c0[r] + bias0);
      float v1 = fast_tanh(c1[r] + bias1);
      *(unsigned*)(hb + r * 32) = packh2(v0, v1);
      sum0 += v0; sq0 += v0 * v0;
      sum1 += v1; sq1 += v1 * v1;
    }
  } else {
#pragma unroll
    for (int r = 0; r < 8; ++r) {
      float v0 = fast_tanh(c0[r] + bias0);
      float v1 = fast_tanh(c1[r] + bias1);
      if ((e0 + r + hi * 8) < E) {
        *(unsigned*)(hb + r * 32) = packh2(v0, v1);
        sum0 += v0; sq0 += v0 * v0;
        sum1 += v1; sq1 += v1 * v1;
      }
    }
  }
  atomicAdd(&s_sum[n], sum0);      atomicAdd(&s_sq[n], sq0);
  atomicAdd(&s_sum[n + 16], sum1); atomicAdd(&s_sq[n + 16], sq1);
}

// =====================================================================
// EdgeConv layer 1 (CIN compile-time): gather [x_i, x_j-x_i], @w1.T+b1,
// tanh, interleaved f16 store, BN stats.
// =====================================================================
template <int CIN>
__global__ void edge_layer1_t(const float* __restrict__ x, int ldx,
                              const int* __restrict__ src, const int* __restrict__ dst,
                              int E, const float* __restrict__ w1,
                              const float* __restrict__ b1,
                              _Float16* __restrict__ hout, float* __restrict__ stats) {
  constexpr int K  = 2 * CIN;
  constexpr int KP = (K + 31) & ~31;
  __shared__ _Float16 s_w[32 * KP];
  __shared__ float s_sum[32];
  __shared__ float s_sq[32];
  int t = threadIdx.x;
  for (int idx = t; idx < 32 * KP; idx += blockDim.x) {
    int row = idx / KP, k = idx % KP;
    s_w[idx] = (_Float16)((k < K) ? w1[row * K + k] : 0.f);
  }
  if (t < 32) { s_sum[t] = 0.f; s_sq[t] = 0.f; }
  __syncthreads();

  int lane = t & 31;
  int tile = uni(blockIdx.x * (blockDim.x >> 5) + (t >> 5));  // SGPR tile id
  int e0 = tile * 16;
  if (e0 < E) {                                   // scalar branch
    int hi = lane >> 4;
    int n = lane & 15;
    int e = e0 + (lane & 15); if (e >= E) e = E - 1;
    const float* xd = x + (long long)dst[e] * ldx;
    const float* xs = x + (long long)src[e] * ldx;

    v8f c0 = zero8f(), c1 = zero8f();
    if constexpr (CIN == 2) {
      v16h a = zero16h();
      if (hi == 0) {
        float t0 = xd[0], t1 = xd[1], s0 = xs[0], s1 = xs[1];
        a[0] = (_Float16)t0;        a[1] = (_Float16)t1;
        a[2] = (_Float16)(s0 - t0); a[3] = (_Float16)(s1 - t1);
      }
      v16h bf0 = ldb16(s_w + n * KP + hi * 8);
      v16h bf1 = ldb16(s_w + (n + 16) * KP + hi * 8);
      c0 = wmma_f16(a, bf0, c0);
      c1 = wmma_f16(a, bf1, c1);
    } else {
      for (int kc = 0; kc < K; kc += 32) {
        int k0 = kc + hi * 8, k1 = kc + 16 + hi * 8;
        f32x4 a0, a1, a2, a3;
        if (kc + 32 <= CIN) {                     // uniform (loop counter)
          a0 = ld4(xd + k0); a1 = ld4(xd + k0 + 4);
          a2 = ld4(xd + k1); a3 = ld4(xd + k1 + 4);
        } else {
          int q0 = k0 - CIN, q1 = k1 - CIN;
          a0 = ld4(xs + q0) - ld4(xd + q0);
          a1 = ld4(xs + q0 + 4) - ld4(xd + q0 + 4);
          a2 = ld4(xs + q1) - ld4(xd + q1);
          a3 = ld4(xs + q1 + 4) - ld4(xd + q1 + 4);
        }
        v16h a = pack16(a0, a1, a2, a3);
        int kb = kc + hi * 8;
        v16h bf0 = ldb16(s_w + n * KP + kb);
        v16h bf1 = ldb16(s_w + (n + 16) * KP + kb);
        c0 = wmma_f16(a, bf0, c0);
        c1 = wmma_f16(a, bf1, c1);
      }
    }
    edge_act_store(c0, c1, b1[n], b1[n + 16], hout, e0, hi, n, E, s_sum, s_sq);
  }
  __syncthreads();
  if (t < 32) {
    atomicAdd(&stats[t], s_sum[t]);
    atomicAdd(&stats[32 + t], s_sq[t]);
  }
}

// =====================================================================
// BN fold: ab[c]=g*rsqrt(var+eps); ab[32+c]=be-mean*ab[c]. <<<1,32>>>
// =====================================================================
__global__ void bn_finalize(const float* __restrict__ stats,
                            const float* __restrict__ g, const float* __restrict__ be,
                            float invE, float* __restrict__ ab) {
  int c = threadIdx.x;
  float m = stats[c] * invE;
  float var = fmaxf(stats[32 + c] * invE - m * m, 0.f);
  float a = g[c] * rsqrtf(var + BN_EPS);
  ab[c] = a;
  ab[32 + c] = be[c] - m * a;
}

// =====================================================================
// EdgeConv layer 2: A=bn(h1), @w2.T(32x32)+b2, tanh, store, stats.
// =====================================================================
__global__ void edge_layer2(const _Float16* __restrict__ hin,
                            const float* __restrict__ ab,
                            const float* __restrict__ w2, const float* __restrict__ b2,
                            int E, _Float16* __restrict__ hout,
                            float* __restrict__ stats) {
  __shared__ _Float16 s_w[32 * 32];
  __shared__ float s_sum[32];
  __shared__ float s_sq[32];
  int t = threadIdx.x;
  for (int idx = t; idx < 32 * 32; idx += blockDim.x)
    s_w[idx] = (_Float16)w2[idx];
  if (t < 32) { s_sum[t] = 0.f; s_sq[t] = 0.f; }
  __syncthreads();

  int lane = t & 31;
  int tile = uni(blockIdx.x * (blockDim.x >> 5) + (t >> 5));
  int e0 = tile * 16;
  if (e0 < E) {                                   // scalar branch
    int hi = lane >> 4;
    int n = lane & 15;
    int e = e0 + (lane & 15); if (e >= E) e = E - 1;

    v16h a = load_bn_a(hin + (long long)e * 32, ab, hi);
    v16h bf0 = ldb16(s_w + n * 32 + hi * 8);
    v16h bf1 = ldb16(s_w + (n + 16) * 32 + hi * 8);
    v8f c0 = zero8f(), c1 = zero8f();
    c0 = wmma_f16(a, bf0, c0);
    c1 = wmma_f16(a, bf1, c1);

    edge_act_store(c0, c1, b2[n], b2[n + 16], hout, e0, hi, n, E, s_sum, s_sq);
  }
  __syncthreads();
  if (t < 32) {
    atomicAdd(&stats[t], s_sum[t]);
    atomicAdd(&stats[32 + t], s_sq[t]);
  }
}

// =====================================================================
// EdgeConv layer 3 (COUT compile-time): A=bn(h2), @w3.T+b3,
// segment-sum scatter via atomicAdd (dst indices broadcast via shfl).
// =====================================================================
template <int COUT>
__global__ void edge_layer3_t(const _Float16* __restrict__ hin,
                              const float* __restrict__ ab,
                              const float* __restrict__ w3, const float* __restrict__ b3,
                              const int* __restrict__ dst, int E,
                              float* __restrict__ out, int ldout) {
  constexpr int NT = (COUT + 15) / 16;
  constexpr int ROWS = NT * 16;
  __shared__ _Float16 s_w[ROWS * 32];
  int t = threadIdx.x;
  for (int idx = t; idx < ROWS * 32; idx += blockDim.x) {
    int row = idx / 32, k = idx % 32;
    s_w[idx] = (_Float16)((row < COUT) ? w3[row * 32 + k] : 0.f);
  }
  __syncthreads();

  int lane = t & 31;
  int tile = uni(blockIdx.x * (blockDim.x >> 5) + (t >> 5));
  int e0 = tile * 16;
  if (e0 >= E) return;                         // scalar branch
  int hi = lane >> 4;
  int n16 = lane & 15;
  int e = e0 + (lane & 15); if (e >= E) e = E - 1;
  int dnode = dst[e];
  long long base[8];
#pragma unroll
  for (int r = 0; r < 8; ++r)
    base[r] = (long long)__shfl(dnode, r + hi * 8, 32) * ldout;

  v16h a = load_bn_a(hin + (long long)e * 32, ab, hi);
#pragma unroll
  for (int nt = 0; nt < NT; ++nt) {
    int row = nt * 16 + n16;
    v16h bf = ldb16(s_w + row * 32 + hi * 8);
    v8f c = zero8f();
    c = wmma_f16(a, bf, c);
    if (row < COUT) {
      float bias = b3[row];
      if (e0 + 16 <= E) {                      // scalar branch
#pragma unroll
        for (int r = 0; r < 8; ++r)
          atomicAdd(out + base[r] + row, c[r] + bias);
      } else {
#pragma unroll
        for (int r = 0; r < 8; ++r)
          if ((e0 + r + hi * 8) < E)
            atomicAdd(out + base[r] + row, c[r] + bias);
      }
    }
  }
}

// =====================================================================
// SAGE: neighbor sum scatter (CIN compile-time: no integer division)
// =====================================================================
template <int CIN>
__global__ void sage_scatter_t(const float* __restrict__ x, int ldx,
                               const int* __restrict__ src,
                               const int* __restrict__ dst,
                               int E, float* __restrict__ agg) {
  long long i = (long long)blockIdx.x * blockDim.x + threadIdx.x;
  long long total = (long long)E * CIN;
  if (i >= total) return;
  int e = (int)(i / CIN);
  int c = (int)(i - (long long)e * CIN);
  atomicAdd(&agg[(long long)dst[e] * CIN + c], x[(long long)src[e] * ldx + c]);
}

__global__ void deg_kernel(const int* __restrict__ dst, int E,
                           float* __restrict__ deg) {
  int e = blockIdx.x * blockDim.x + threadIdx.x;
  if (e < E) atomicAdd(&deg[dst[e]], 1.f);
}

// =====================================================================
// SAGE node update (CIN/COUT/LEAKY compile-time; out stride == COUT):
// out = mean @ wl.T + bl + x @ wr.T  (two WMMAs per K-chunk per N-tile)
// =====================================================================
template <int CIN, int COUT, bool LEAKY>
__global__ void sage_node_t(const float* __restrict__ x, int ldx,
                            const float* __restrict__ agg,
                            const float* __restrict__ deg,
                            const float* __restrict__ wl,
                            const float* __restrict__ bl,
                            const float* __restrict__ wr,
                            int Nn, float* __restrict__ out) {
  constexpr int KP = (CIN + 31) & ~31;
  constexpr int NT = (COUT + 15) / 16;
  constexpr int ROWS = NT * 16;
  __shared__ _Float16 s_wl[ROWS * KP];
  __shared__ _Float16 s_wr[ROWS * KP];
  int t = threadIdx.x;
  for (int idx = t; idx < ROWS * KP; idx += blockDim.x) {
    int row = idx / KP, k = idx % KP;
    float vl = 0.f, vr = 0.f;
    if (row < COUT && k < CIN) { vl = wl[row * CIN + k]; vr = wr[row * CIN + k]; }
    s_wl[idx] = (_Float16)vl;
    s_wr[idx] = (_Float16)vr;
  }
  __syncthreads();

  int lane = t & 31;
  int tile = uni(blockIdx.x * (blockDim.x >> 5) + (t >> 5));
  int v0i = tile * 16;
  if (v0i >= Nn) return;                      // scalar branch
  int hi = lane >> 4;
  int n16 = lane & 15;
  int nd = v0i + (lane & 15); if (nd >= Nn) nd = Nn - 1;
  float rd = 1.f / fmaxf(deg[nd], 1.f);
  const float* xr = x + (long long)nd * ldx;
  const float* ar = agg + (long long)nd * CIN;

  v8f acc[NT];
#pragma unroll
  for (int i = 0; i < NT; ++i) acc[i] = zero8f();

#pragma unroll
  for (int kc = 0; kc < KP; kc += 32) {
    v16h am, ax;
    if constexpr (CIN == 3) {
      am = zero16h(); ax = zero16h();
      if (hi == 0) {
        am[0] = (_Float16)(ar[0] * rd); am[1] = (_Float16)(ar[1] * rd);
        am[2] = (_Float16)(ar[2] * rd);
        ax[0] = (_Float16)xr[0]; ax[1] = (_Float16)xr[1]; ax[2] = (_Float16)xr[2];
      }
    } else {
      int k0 = kc + hi * 8, k1 = kc + 16 + hi * 8;
      am = pack16(ld4(ar + k0) * rd, ld4(ar + k0 + 4) * rd,
                  ld4(ar + k1) * rd, ld4(ar + k1 + 4) * rd);
      ax = pack16(ld4(xr + k0), ld4(xr + k0 + 4),
                  ld4(xr + k1), ld4(xr + k1 + 4));
    }
    int kb = kc + hi * 8;
#pragma unroll
    for (int nt = 0; nt < NT; ++nt) {
      int row = nt * 16 + n16;
      v16h bL = ldb16(s_wl + row * KP + kb);
      v16h bR = ldb16(s_wr + row * KP + kb);
      acc[nt] = wmma_f16(am, bL, acc[nt]);
      acc[nt] = wmma_f16(ax, bR, acc[nt]);
    }
  }
  // out stride == COUT; base for this lane's row group
  float* ob = out + (long long)(v0i + hi * 8) * COUT;
#pragma unroll
  for (int nt = 0; nt < NT; ++nt) {
    int row = nt * 16 + n16;
    if (row < COUT) {
      float bias = bl[row];
      if (v0i + 16 <= Nn) {                    // scalar branch
#pragma unroll
        for (int r = 0; r < 8; ++r) {
          float v = acc[nt][r] + bias;
          if (LEAKY) v = (v > 0.f) ? v : 0.1f * v;
          ob[r * COUT + row] = v;
        }
      } else {
#pragma unroll
        for (int r = 0; r < 8; ++r) {
          if ((v0i + r + hi * 8) < Nn) {
            float v = acc[nt][r] + bias;
            if (LEAKY) v = (v > 0.f) ? v : 0.1f * v;
            ob[r * COUT + row] = v;
          }
        }
      }
    }
  }
}

__global__ void add_vec(const float* __restrict__ a, float* __restrict__ io, int n) {
  int i = blockIdx.x * blockDim.x + threadIdx.x;
  if (i < n) io[i] = a[i] + io[i];
}

// =====================================================================
// Host orchestration
// =====================================================================
struct ECP { const float *w1, *b1, *g1, *be1, *w2, *b2, *g2, *be2, *w3, *b3; };
struct SP  { const float *wl, *bl, *wr; };

template <int CIN, int COUT>
static void run_edge_conv(const float* x, int ldx, float* out, int ldout,
                          const ECP& q, const int* src, const int* dst, int E,
                          _Float16* h1, _Float16* h2, float* stats,
                          float* ab1, float* ab2, dim3 egrd, dim3 eblk,
                          float invE, hipStream_t stream) {
  hipMemsetAsync(stats, 0, 64 * 4, stream);
  edge_layer1_t<CIN><<<egrd, eblk, 0, stream>>>(x, ldx, src, dst, E,
                                                q.w1, q.b1, h1, stats);
  bn_finalize<<<1, 32, 0, stream>>>(stats, q.g1, q.be1, invE, ab1);
  hipMemsetAsync(stats, 0, 64 * 4, stream);
  edge_layer2<<<egrd, eblk, 0, stream>>>(h1, ab1, q.w2, q.b2, E, h2, stats);
  bn_finalize<<<1, 32, 0, stream>>>(stats, q.g2, q.be2, invE, ab2);
  edge_layer3_t<COUT><<<egrd, eblk, 0, stream>>>(h2, ab2, q.w3, q.b3, dst, E,
                                                 out, ldout);
}

template <int CIN, int COUT, bool LEAKY>
static void run_sage(const float* x, int ldx, const SP& q, float* out,
                     const int* src, const int* dst, int E, int N,
                     float* agg, float* deg, hipStream_t stream) {
  hipMemsetAsync(agg, 0, (size_t)N * CIN * 4, stream);
  long long tot = (long long)E * CIN;
  int sblocks = (int)((tot + 255) / 256);
  sage_scatter_t<CIN><<<sblocks, 256, 0, stream>>>(x, ldx, src, dst, E, agg);
  int ntl = (N + 15) / 16;
  sage_node_t<CIN, COUT, LEAKY><<<(ntl + WAVES_PER_BLOCK - 1) / WAVES_PER_BLOCK,
                                  EDGE_BLOCK, 0, stream>>>(
      x, ldx, agg, deg, q.wl, q.bl, q.wr, N, out);
}

extern "C" void kernel_launch(void* const* d_in, const int* in_sizes, int n_in,
                              void* d_out, int out_size, void* d_ws, size_t ws_size,
                              hipStream_t stream) {
  (void)out_size; (void)ws_size;
  // ---- input index resolution (dict-insertion vs pytree-sorted order) ----
  bool sorted_top = (in_sizes[0] != 300000);   // insertion: u (100000*3) first
  int iu, icoord, iedge, ip;
  if (!sorted_top) { iu = 0; icoord = 1; iedge = 2; ip = 3; }
  else             { icoord = 0; iedge = 1; ip = 2; iu = n_in - 1; }

  const float* u     = (const float*)d_in[iu];
  const float* coord = (const float*)d_in[icoord];
  const int*   ei    = (const int*)d_in[iedge];
  int E = in_sizes[iedge] / 2;
  int N = in_sizes[iu] / 3;
  const int* src = ei;
  const int* dst = ei + E;

  auto getf = [&](int i) { return (const float*)d_in[i]; };
  auto parse_ec = [&](int& p) -> ECP {
    ECP q;
    bool sk = (in_sizes[p] == 32);   // sorted keys start with b1 (size 32)
    if (!sk) {
      q.w1 = getf(p);   q.b1 = getf(p+1); q.g1 = getf(p+2); q.be1 = getf(p+3);
      q.w2 = getf(p+4); q.b2 = getf(p+5); q.g2 = getf(p+6); q.be2 = getf(p+7);
      q.w3 = getf(p+8); q.b3 = getf(p+9);
    } else { // b1,b2,b3,be1,be2,g1,g2,w1,w2,w3
      q.b1 = getf(p);    q.b2 = getf(p+1); q.b3 = getf(p+2); q.be1 = getf(p+3);
      q.be2 = getf(p+4); q.g1 = getf(p+5); q.g2 = getf(p+6); q.w1 = getf(p+7);
      q.w2 = getf(p+8);  q.w3 = getf(p+9);
    }
    p += 10; return q;
  };
  auto parse_sp = [&](int& p, int cout) -> SP {
    SP q;
    bool sk = (in_sizes[p] == cout); // sorted keys start with bl (size cout)
    if (!sk) { q.wl = getf(p); q.bl = getf(p+1); q.wr = getf(p+2); }
    else     { q.bl = getf(p); q.wl = getf(p+1); q.wr = getf(p+2); }
    p += 3; return q;
  };

  ECP ec1, ecs[3], ec2, ec3; SP sg4, sgs[3], sg5;
  {
    int p = ip;
    if (!sorted_top) {
      ec1 = parse_ec(p);
      for (int i = 0; i < 3; ++i) ecs[i] = parse_ec(p);
      ec2 = parse_ec(p); ec3 = parse_ec(p);
      sg4 = parse_sp(p, 64);
      for (int i = 0; i < 3; ++i) sgs[i] = parse_sp(p, 64);
      sg5 = parse_sp(p, 3);
    } else { // ec1, ec2, ec3, ecs, sage4, sage5, sages
      ec1 = parse_ec(p); ec2 = parse_ec(p); ec3 = parse_ec(p);
      for (int i = 0; i < 3; ++i) ecs[i] = parse_ec(p);
      sg4 = parse_sp(p, 64); sg5 = parse_sp(p, 3);
      for (int i = 0; i < 3; ++i) sgs[i] = parse_sp(p, 64);
    }
  }

  // ---- workspace carve ----
  char* wsp = (char*)d_ws;
  size_t off = 0;
  auto carve = [&](size_t bytes) -> void* {
    void* r = wsp + off; off += (bytes + 255) & ~(size_t)255; return r;
  };
  float*    append = (float*)carve((size_t)N * 256 * 4);
  float*    x2     = (float*)carve((size_t)N * 64 * 4);
  float*    x3     = (float*)carve((size_t)N * 3 * 4);
  float*    hA     = (float*)carve((size_t)N * 64 * 4);
  float*    hB     = (float*)carve((size_t)N * 64 * 4);
  float*    agg    = (float*)carve((size_t)N * 64 * 4);
  float*    deg    = (float*)carve((size_t)N * 4);
  _Float16* h1     = (_Float16*)carve((size_t)E * 32 * 2);
  _Float16* h2     = (_Float16*)carve((size_t)E * 32 * 2);
  float*    stats  = (float*)carve(64 * 4);
  float*    ab1    = (float*)carve(64 * 4);
  float*    ab2    = (float*)carve(64 * 4);

  int etiles = (E + 15) / 16;
  dim3 eblk(EDGE_BLOCK);
  dim3 egrd((etiles + WAVES_PER_BLOCK - 1) / WAVES_PER_BLOCK);
  float invE = 1.f / (float)E;

  // ---- EdgeConv stack (append = concat of four conv outputs) ----
  hipMemsetAsync(append, 0, (size_t)N * 256 * 4, stream);
  run_edge_conv<2, 64>(coord, 2, append + 0, 256, ec1, src, dst, E,
                       h1, h2, stats, ab1, ab2, egrd, eblk, invE, stream);
  run_edge_conv<64, 64>(append + 0, 256, append + 64, 256, ecs[0], src, dst, E,
                        h1, h2, stats, ab1, ab2, egrd, eblk, invE, stream);
  run_edge_conv<64, 64>(append + 64, 256, append + 128, 256, ecs[1], src, dst, E,
                        h1, h2, stats, ab1, ab2, egrd, eblk, invE, stream);
  run_edge_conv<64, 64>(append + 128, 256, append + 192, 256, ecs[2], src, dst, E,
                        h1, h2, stats, ab1, ab2, egrd, eblk, invE, stream);
  hipMemsetAsync(x2, 0, (size_t)N * 64 * 4, stream);
  run_edge_conv<256, 64>(append, 256, x2, 64, ec2, src, dst, E,
                         h1, h2, stats, ab1, ab2, egrd, eblk, invE, stream);
  hipMemsetAsync(x3, 0, (size_t)N * 3 * 4, stream);
  run_edge_conv<64, 3>(x2, 64, x3, 3, ec3, src, dst, E,
                       h1, h2, stats, ab1, ab2, egrd, eblk, invE, stream);

  // h0 = u + x3  (in place in x3)
  add_vec<<<(N * 3 + 255) / 256, 256, 0, stream>>>(u, x3, N * 3);

  // degrees (shared by all SAGE layers)
  hipMemsetAsync(deg, 0, (size_t)N * 4, stream);
  deg_kernel<<<(E + 255) / 256, 256, 0, stream>>>(dst, E, deg);

  // ---- SAGE stack ----
  run_sage<3, 64, true>(x3, 3, sg4, hA, src, dst, E, N, agg, deg, stream);
  run_sage<64, 64, true>(hA, 64, sgs[0], hB, src, dst, E, N, agg, deg, stream);
  run_sage<64, 64, true>(hB, 64, sgs[1], hA, src, dst, E, N, agg, deg, stream);
  run_sage<64, 64, true>(hA, 64, sgs[2], hB, src, dst, E, N, agg, deg, stream);
  run_sage<64, 3, false>(hB, 64, sg5, (float*)d_out, src, dst, E, N, agg, deg,
                         stream);
}
